// BiLSTM_CRF_14980845928896
// MI455X (gfx1250) — compile-verified, matrix operand order checked by reading, not run on hardware
//
#include <hip/hip_runtime.h>
#include <hip/hip_bf16.h>

// ---------------------------------------------------------------------------
// BiLSTM-CRF for MI455X (gfx1250, wave32)
//   T=65536 timesteps, E=300 embed, H=64 per-direction hidden, NT=3 tags
//
// Phase 1: input GEMMs  G = X @ W_ih^T + (b_ih+b_hh)   via v_wmma_f32_16x16x32_f16
// Phase 2: sequential LSTM recurrence, both directions concurrently (2 blocks),
//          register-pipelined gate loads + global_prefetch streaming
// Phase 3: emissions  emis = [hF|hB] @ W_tag^T + b_tag
// Phase 4: CRF as chunked associative scans (lse-semiring / max-plus / map
//          composition), 256 chunks x 256 steps -> ~100x shorter critical path
// ---------------------------------------------------------------------------

#define TT   65536
#define EE   300
#define HH   64
#define G4H  256   // 4*H
#define NTAG 3
#define NCH  256   // CRF chunks
#define CHL  256   // CRF chunk length (transitions per chunk)

typedef __attribute__((ext_vector_type(16))) _Float16 v16h;
typedef __attribute__((ext_vector_type(8)))  float    v8f;

// ---------------------------------------------------------------------------
// small helpers (register-only dynamic selects; no scratch-forcing indexing)
// ---------------------------------------------------------------------------
__device__ __forceinline__ float lse3(float a, float b, float c) {
    const float m = fmaxf(a, fmaxf(b, c));
    return m + __logf(__expf(a - m) + __expf(b - m) + __expf(c - m));
}
__device__ __forceinline__ float pick3f(float a, float b, float c, int i) {
    return (i == 0) ? a : ((i == 1) ? b : c);
}
__device__ __forceinline__ int pick3i(int a, int b, int c, int i) {
    return (i == 0) ? a : ((i == 1) ? b : c);
}
// tr[i][j] with dynamic i,j resolved via cndmask chains (constant array refs)
__device__ __forceinline__ float trsel(const float tr[3][3], int i, int j) {
    const float r0 = pick3f(tr[0][0], tr[0][1], tr[0][2], j);
    const float r1 = pick3f(tr[1][0], tr[1][1], tr[1][2], j);
    const float r2 = pick3f(tr[2][0], tr[2][1], tr[2][2], j);
    return pick3f(r0, r1, r2, i);
}

// ---------------------------------------------------------------------------
// Phase 1: G[T,256] = X[T,300] @ W[256,300]^T + (bih+bhh)
// One wave = one 16x16 output tile; K in 10 chunks of 32 (pad 300->320).
// ---------------------------------------------------------------------------
__global__ __launch_bounds__(256)
void lstm_input_gemm(const float* __restrict__ X,     // [T, 300]
                     const float* __restrict__ W,     // [256, 300] row-major
                     const float* __restrict__ bih,   // [256]
                     const float* __restrict__ bhh,   // [256]
                     float* __restrict__ G)           // [T, 256]
{
    const int lane = threadIdx.x & 31;
    const int wave = threadIdx.x >> 5;
    const int m0   = (blockIdx.x * 8 + wave) * 16;   // row tile of X / G
    const int n0   = blockIdx.y * 16;                // col tile of G (gate index)
    const int row  = lane & 15;                      // A: matrix row; B/D: column
    const int hi   = lane >> 4;

    v8f acc = {};
    #pragma unroll
    for (int kk = 0; kk < 10; ++kk) {
        const int kbase = kk * 32;
        v16h a, b;
        #pragma unroll
        for (int e = 0; e < 16; ++e) {
            // CDNA5 16-bit A 16x32 fragment: K = (e&7) + (e>>3)*16 + hi*8
            const int k = kbase + (e & 7) + ((e >> 3) << 4) + hi * 8;
            float av = 0.0f, bv = 0.0f;
            if (k < EE) {
                av = X[(size_t)(m0 + row) * EE + k];
                bv = W[(size_t)(n0 + row) * EE + k];   // column n0+row of W^T
            }
            a[e] = (_Float16)av;
            b[e] = (_Float16)bv;
        }
        acc = __builtin_amdgcn_wmma_f32_16x16x32_f16(
            /*neg_a=*/false, a, /*neg_b=*/false, b,
            /*c_mod=*/(short)0, acc, /*reuse_a=*/false, /*reuse_b=*/false);
    }

    // D layout: N = lane&15, M = r + (lane>>4)*8
    const float bias = bih[n0 + row] + bhh[n0 + row];
    #pragma unroll
    for (int r = 0; r < 8; ++r) {
        const int m = m0 + r + hi * 8;
        G[(size_t)m * G4H + (n0 + row)] = acc[r] + bias;
    }
}

// ---------------------------------------------------------------------------
// Phase 2: sequential recurrence. blockIdx.x = 0 (forward) / 1 (backward).
// Thread j owns gate j (W_hh row in 64 VGPRs); h broadcast via LDS.
// Gate pre-activation loads are register-pipelined one step ahead so the
// global-load latency hides behind the 64-FMA dot chain + barriers.
// ---------------------------------------------------------------------------
__global__ __launch_bounds__(256)
void lstm_recurrence(const float* __restrict__ GF, const float* __restrict__ GB,
                     const float* __restrict__ WhhF, const float* __restrict__ WhhB,
                     float* __restrict__ HF, float* __restrict__ HB)
{
    const int dir = blockIdx.x;                     // 0 = fwd, 1 = bwd
    const float* __restrict__ G    = dir ? GB   : GF;
    const float* __restrict__ Whh  = dir ? WhhB : WhhF;
    float*       __restrict__ Hout = dir ? HB   : HF;

    __shared__ float h_sh[HH];
    __shared__ float g_sh[G4H];

    const int j = threadIdx.x;

    float w[HH];
    #pragma unroll
    for (int q = 0; q < HH; ++q) w[q] = Whh[(size_t)j * HH + q];

    float c = 0.0f;
    if (j < HH) h_sh[j] = 0.0f;
    __syncthreads();

    const int t0 = dir ? (TT - 1) : 0;
    float gnext = G[(size_t)t0 * G4H + j];          // pipelined load

    for (int s = 0; s < TT; ++s) {
        const int t = dir ? (TT - 1 - s) : s;
        const float gcur = gnext;

        if (s + 1 < TT) {
            const int tn = dir ? (t - 1) : (t + 1);
            gnext = G[(size_t)tn * G4H + j];        // issue early for next step
            const int tp = dir ? (t - 16) : (t + 16);
            if (tp >= 0 && tp < TT)
                __builtin_prefetch(&G[(size_t)tp * G4H + j], 0, 1);  // global_prefetch
        }

        float acc = gcur;
        #pragma unroll
        for (int q = 0; q < HH; ++q) acc = fmaf(w[q], h_sh[q], acc);
        g_sh[j] = acc;
        __syncthreads();             // dots done; g_sh complete; h_sh free

        if (j < HH) {
            const float gi = 1.0f / (1.0f + __expf(-g_sh[j]));
            const float gf = 1.0f / (1.0f + __expf(-g_sh[j + HH]));
            const float gg = g_sh[j + 2 * HH];
            const float go = 1.0f / (1.0f + __expf(-g_sh[j + 3 * HH]));
            c = gf * c + gi * tanhf(gg);
            const float h = go * tanhf(c);
            h_sh[j] = h;
            Hout[(size_t)t * HH + j] = h;
        }
        __syncthreads();             // h_sh published for next step
    }
}

// ---------------------------------------------------------------------------
// Phase 3: emis[t,k] = hF[t,:]*Wtag[k,:64] + hB[t,:]*Wtag[k,64:] + btag[k]
// ---------------------------------------------------------------------------
__global__ __launch_bounds__(256)
void emission_kernel(const float* __restrict__ HF, const float* __restrict__ HB,
                     const float* __restrict__ Wtag,   // [3, 128]
                     const float* __restrict__ btag,   // [3]
                     float* __restrict__ emis)         // [T, 3]
{
    __shared__ float wt[NTAG * 2 * HH];
    __shared__ float bt[NTAG];
    for (int i = threadIdx.x; i < NTAG * 2 * HH; i += blockDim.x) wt[i] = Wtag[i];
    if (threadIdx.x < NTAG) bt[threadIdx.x] = btag[threadIdx.x];
    __syncthreads();

    const int t = blockIdx.x * blockDim.x + threadIdx.x;
    if (t >= TT) return;

    float e0 = bt[0], e1 = bt[1], e2 = bt[2];
    #pragma unroll 4
    for (int q = 0; q < HH; ++q) {
        const float v = HF[(size_t)t * HH + q];
        e0 = fmaf(v, wt[0 * 2 * HH + q], e0);
        e1 = fmaf(v, wt[1 * 2 * HH + q], e1);
        e2 = fmaf(v, wt[2 * 2 * HH + q], e2);
    }
    #pragma unroll 4
    for (int q = 0; q < HH; ++q) {
        const float v = HB[(size_t)t * HH + q];
        e0 = fmaf(v, wt[0 * 2 * HH + HH + q], e0);
        e1 = fmaf(v, wt[1 * 2 * HH + HH + q], e1);
        e2 = fmaf(v, wt[2 * 2 * HH + HH + q], e2);
    }
    emis[(size_t)t * NTAG + 0] = e0;
    emis[(size_t)t * NTAG + 1] = e1;
    emis[(size_t)t * NTAG + 2] = e2;
}

// ---------------------------------------------------------------------------
// Phase 4a: per-chunk scans (parallel, 256 chunks).
// Chunk ch covers transitions t in [1+ch*CHL, min(T-1, (ch+1)*CHL)].
// Produces: lse-semiring 3x3 product M, max-plus 3x3 product V, score partial.
// ---------------------------------------------------------------------------
__global__ void crf_chunk_scan(const float* __restrict__ emis,
                               const int*   __restrict__ tags,
                               const float* __restrict__ trans,
                               float* __restrict__ Mch,   // [NCH][9]
                               float* __restrict__ Vch,   // [NCH][9]
                               float* __restrict__ Sch)   // [NCH]
{
    const int ch = blockIdx.x * blockDim.x + threadIdx.x;
    if (ch >= NCH) return;

    float tr[3][3];
    #pragma unroll
    for (int a = 0; a < 3; ++a)
        #pragma unroll
        for (int b = 0; b < 3; ++b) tr[a][b] = trans[a * 3 + b];

    const int t_lo = 1 + ch * CHL;
    const int t_hi = min(TT - 1, t_lo + CHL - 1);

    // init with first transition of the chunk
    float e0 = emis[(size_t)t_lo * NTAG + 0];
    float e1 = emis[(size_t)t_lo * NTAG + 1];
    float e2 = emis[(size_t)t_lo * NTAG + 2];
    float P[3][3], V[3][3];
    #pragma unroll
    for (int i = 0; i < 3; ++i)
        #pragma unroll
        for (int jj = 0; jj < 3; ++jj) {
            const float v = tr[i][jj] + pick3f(e0, e1, e2, jj);
            P[i][jj] = v; V[i][jj] = v;
        }
    int   pt = tags[t_lo - 1];
    int   tg = tags[t_lo];
    float sc = pick3f(e0, e1, e2, tg) + trsel(tr, pt, tg);

    for (int t = t_lo + 1; t <= t_hi; ++t) {
        e0 = emis[(size_t)t * NTAG + 0];
        e1 = emis[(size_t)t * NTAG + 1];
        e2 = emis[(size_t)t * NTAG + 2];
        pt = tg;
        tg = tags[t];
        sc += pick3f(e0, e1, e2, tg) + trsel(tr, pt, tg);

        float nP[3][3], nV[3][3];
        #pragma unroll
        for (int i = 0; i < 3; ++i)
            #pragma unroll
            for (int jj = 0; jj < 3; ++jj) {
                const float ej = pick3f(e0, e1, e2, jj);
                nP[i][jj] = lse3(P[i][0] + tr[0][jj],
                                 P[i][1] + tr[1][jj],
                                 P[i][2] + tr[2][jj]) + ej;
                nV[i][jj] = fmaxf(V[i][0] + tr[0][jj],
                           fmaxf(V[i][1] + tr[1][jj],
                                 V[i][2] + tr[2][jj])) + ej;
            }
        #pragma unroll
        for (int i = 0; i < 3; ++i)
            #pragma unroll
            for (int jj = 0; jj < 3; ++jj) { P[i][jj] = nP[i][jj]; V[i][jj] = nV[i][jj]; }
    }

    #pragma unroll
    for (int i = 0; i < 3; ++i)
        #pragma unroll
        for (int jj = 0; jj < 3; ++jj) {
            Mch[(size_t)ch * 9 + i * 3 + jj] = P[i][jj];
            Vch[(size_t)ch * 9 + i * 3 + jj] = V[i][jj];
        }
    Sch[ch] = sc;
}

// ---------------------------------------------------------------------------
// Phase 4b: serial combine over 256 chunks (tiny). Emits loss, Viterbi
// boundary vectors per chunk, and the terminal Viterbi tag.
// ---------------------------------------------------------------------------
__global__ void crf_combine(const float* __restrict__ emis,
                            const int*   __restrict__ tags,
                            const float* __restrict__ startv,
                            const float* __restrict__ endv,
                            const float* __restrict__ Mch,
                            const float* __restrict__ Vch,
                            const float* __restrict__ Sch,
                            float*    __restrict__ svB,      // [NCH][3]
                            unsigned* __restrict__ lastTag,  // [1]
                            float*    __restrict__ out)
{
    if (threadIdx.x != 0 || blockIdx.x != 0) return;

    float alpha[3], sv[3];
    #pragma unroll
    for (int k = 0; k < 3; ++k) {
        alpha[k] = startv[k] + emis[k];
        sv[k]    = alpha[k];
    }
    const int tg0 = tags[0];
    float score = pick3f(startv[0], startv[1], startv[2], tg0) +
                  pick3f(emis[0], emis[1], emis[2], tg0);

    for (int ch = 0; ch < NCH; ++ch) {
        svB[ch * 3 + 0] = sv[0];
        svB[ch * 3 + 1] = sv[1];
        svB[ch * 3 + 2] = sv[2];
        float na[3], ns[3];
        #pragma unroll
        for (int jj = 0; jj < 3; ++jj) {
            na[jj] = lse3(alpha[0] + Mch[(size_t)ch * 9 + 0 * 3 + jj],
                          alpha[1] + Mch[(size_t)ch * 9 + 1 * 3 + jj],
                          alpha[2] + Mch[(size_t)ch * 9 + 2 * 3 + jj]);
            ns[jj] = fmaxf(sv[0] + Vch[(size_t)ch * 9 + 0 * 3 + jj],
                    fmaxf(sv[1] + Vch[(size_t)ch * 9 + 1 * 3 + jj],
                          sv[2] + Vch[(size_t)ch * 9 + 2 * 3 + jj]));
        }
        #pragma unroll
        for (int k = 0; k < 3; ++k) { alpha[k] = na[k]; sv[k] = ns[k]; }
        score += Sch[ch];
    }

    const float logZ = lse3(alpha[0] + endv[0], alpha[1] + endv[1], alpha[2] + endv[2]);
    const int tgl = tags[TT - 1];
    score += pick3f(endv[0], endv[1], endv[2], tgl);
    out[0] = -(score - logZ) * 10.0f;

    const float f0 = sv[0] + endv[0];
    const float f1 = sv[1] + endv[1];
    const float f2 = sv[2] + endv[2];
    int last = 0; float fm = f0;
    if (f1 > fm) { fm = f1; last = 1; }
    if (f2 > fm) { fm = f2; last = 2; }
    lastTag[0] = (unsigned)last;
}

// ---------------------------------------------------------------------------
// Phase 4c: per-chunk backpointer regeneration + 6-bit traceback composition.
// ---------------------------------------------------------------------------
__global__ void crf_chunk_bp(const float* __restrict__ emis,
                             const float* __restrict__ trans,
                             const float* __restrict__ svB,
                             unsigned* __restrict__ bp,    // [T] packed 2-bit x3
                             unsigned* __restrict__ gpk)   // [NCH] packed maps
{
    const int ch = blockIdx.x * blockDim.x + threadIdx.x;
    if (ch >= NCH) return;

    float tr[3][3];
    #pragma unroll
    for (int a = 0; a < 3; ++a)
        #pragma unroll
        for (int b = 0; b < 3; ++b) tr[a][b] = trans[a * 3 + b];

    const int t_lo = 1 + ch * CHL;
    const int t_hi = min(TT - 1, t_lo + CHL - 1);

    float sv0 = svB[ch * 3 + 0], sv1 = svB[ch * 3 + 1], sv2 = svB[ch * 3 + 2];
    int g0 = 0, g1 = 1, g2 = 2;   // identity map tag@t -> tag@(t_lo-1)

    for (int t = t_lo; t <= t_hi; ++t) {
        const float e0 = emis[(size_t)t * NTAG + 0];
        const float e1 = emis[(size_t)t * NTAG + 1];
        const float e2 = emis[(size_t)t * NTAG + 2];

        unsigned code = 0u;
        float ns[3]; int am[3];
        #pragma unroll
        for (int k = 0; k < 3; ++k) {
            const float y0 = sv0 + tr[0][k];
            const float y1 = sv1 + tr[1][k];
            const float y2 = sv2 + tr[2][k];
            int a = 0; float ym = y0;                 // first-max tie-break
            if (y1 > ym) { ym = y1; a = 1; }
            if (y2 > ym) { ym = y2; a = 2; }
            am[k] = a;
            ns[k] = ym + pick3f(e0, e1, e2, k);
            code |= (unsigned)a << (2 * k);
        }
        bp[t] = code;
        sv0 = ns[0]; sv1 = ns[1]; sv2 = ns[2];
        // G_t(x) = G_{t-1}(bp_t(x))
        const int n0 = pick3i(g0, g1, g2, am[0]);
        const int n1 = pick3i(g0, g1, g2, am[1]);
        const int n2 = pick3i(g0, g1, g2, am[2]);
        g0 = n0; g1 = n1; g2 = n2;
    }
    gpk[ch] = (unsigned)g0 | ((unsigned)g1 << 2) | ((unsigned)g2 << 4);
}

// ---------------------------------------------------------------------------
// Phase 4d: resolve chunk-boundary tags (serial over 256 maps), then fill
// all predictions in parallel (one chunk per thread).
// ---------------------------------------------------------------------------
__global__ __launch_bounds__(256)
void crf_traceback(const unsigned* __restrict__ bp,
                   const unsigned* __restrict__ gpk,
                   const unsigned* __restrict__ lastTag,
                   float* __restrict__ out)
{
    __shared__ int top[NCH];   // tag at position t_hi(ch)
    if (threadIdx.x == 0) {
        int cur = (int)lastTag[0];
        out[1 + (TT - 1)] = (float)cur;
        for (int ch = NCH - 1; ch >= 0; --ch) {
            top[ch] = cur;
            cur = (int)((gpk[ch] >> (2 * cur)) & 3u);
        }
    }
    __syncthreads();

    const int ch = threadIdx.x;
    const int t_lo = 1 + ch * CHL;
    const int t_hi = min(TT - 1, t_lo + CHL - 1);
    int cur = top[ch];
    for (int t = t_hi; t >= t_lo; --t) {
        cur = (int)((bp[t] >> (2 * cur)) & 3u);
        out[1 + (t - 1)] = (float)cur;
    }
}

// ---------------------------------------------------------------------------
extern "C" void kernel_launch(void* const* d_in, const int* in_sizes, int n_in,
                              void* d_out, int out_size, void* d_ws, size_t ws_size,
                              hipStream_t stream)
{
    const float* X      = (const float*)d_in[0];   // [1,T,E]
    const int*   tags   = (const int*)  d_in[1];   // [1,T]
    const float* WihF   = (const float*)d_in[2];
    const float* WhhF   = (const float*)d_in[3];
    const float* bihF   = (const float*)d_in[4];
    const float* bhhF   = (const float*)d_in[5];
    const float* WihB   = (const float*)d_in[6];
    const float* WhhB   = (const float*)d_in[7];
    const float* bihB   = (const float*)d_in[8];
    const float* bhhB   = (const float*)d_in[9];
    const float* Wtag   = (const float*)d_in[10];
    const float* btag   = (const float*)d_in[11];
    const float* cstart = (const float*)d_in[12];
    const float* ctrans = (const float*)d_in[13];
    const float* cend   = (const float*)d_in[14];
    float* out = (float*)d_out;

    // Workspace layout (~161 MB)
    float*    gatesF = (float*)d_ws;                          // T*256
    float*    gatesB = gatesF + (size_t)TT * G4H;             // T*256
    float*    hF     = gatesB + (size_t)TT * G4H;             // T*64
    float*    hB     = hF     + (size_t)TT * HH;              // T*64
    float*    emis   = hB     + (size_t)TT * HH;              // T*3
    unsigned* bps    = (unsigned*)(emis + (size_t)TT * NTAG); // T packed bps
    float*    Mch    = (float*)(bps + (size_t)TT);            // NCH*9
    float*    Vch    = Mch + (size_t)NCH * 9;                 // NCH*9
    float*    Sch    = Vch + (size_t)NCH * 9;                 // NCH
    float*    svB    = Sch + NCH;                             // NCH*3
    unsigned* gpk    = (unsigned*)(svB + (size_t)NCH * 3);    // NCH
    unsigned* lastT  = gpk + NCH;                             // 1

    // Phase 1: both input GEMMs (WMMA). grid: 512 x 16, 8 waves/block.
    dim3 ggrid(TT / (16 * 8), G4H / 16, 1);
    lstm_input_gemm<<<ggrid, 256, 0, stream>>>(X, WihF, bihF, bhhF, gatesF);
    lstm_input_gemm<<<ggrid, 256, 0, stream>>>(X, WihB, bihB, bhhB, gatesB);

    // Phase 2: both directions concurrently.
    lstm_recurrence<<<2, 256, 0, stream>>>(gatesF, gatesB, WhhF, WhhB, hF, hB);

    // Phase 3: emissions.
    emission_kernel<<<TT / 256, 256, 0, stream>>>(hF, hB, Wtag, btag, emis);

    // Phase 4: CRF chunked scans.
    crf_chunk_scan<<<NCH / 32, 32, 0, stream>>>(emis, tags, ctrans, Mch, Vch, Sch);
    crf_combine<<<1, 32, 0, stream>>>(emis, tags, cstart, cend, Mch, Vch, Sch,
                                      svB, lastT, out);
    crf_chunk_bp<<<NCH / 32, 32, 0, stream>>>(emis, ctrans, svB, bps, gpk);
    crf_traceback<<<1, 256, 0, stream>>>(bps, gpk, lastT, out);
}
